// MissingvalueModel_7026566496330
// MI455X (gfx1250) — matrix operand order repaired and checked by reading, not compiled, not run on hardware
//
#include <hip/hip_runtime.h>
#include <hip/hip_bf16.h>

// ---------------------------------------------------------------------------
// GRU-D model for MI455X (gfx1250). wave32 + v_wmma_f32_16x16x32_bf16.
// Strategy: hoist all h-independent GEMMs out of the recurrences into big
// WMMA GEMMs; keep hidden state in LDS (16 batch rows / workgroup); stream
// pre-packed bf16 weights from L2 inside the 512-step scans.
// ---------------------------------------------------------------------------

typedef __attribute__((ext_vector_type(16))) __bf16 v16bf;
typedef __attribute__((ext_vector_type(8)))  __bf16 v8bf;
typedef __attribute__((ext_vector_type(8)))  float  v8f;

#define BB 128
#define TT 512
#define FF 128
#define HH 256
#define EE 256
#define NBT (BB*TT)          // 65536 rows

__device__ __forceinline__ float sigf(float x) { return 1.f / (1.f + __expf(-x)); }

// A-fragment (16x32 bf16, M across lanes-mod-16): K contiguous per half-wave,
// lanes 0-15 hold K 0..15, lanes 16-31 hold K 16..31 -> one 32B load.
__device__ __forceinline__ v16bf ldfragA(const __bf16* base, int hs) {
  return *(const v16bf*)(base + (hs << 4));
}
// B-fragment (32x16 bf16, N across lanes-mod-16, K contiguous per half-wave):
// one 32B load from a [N][K]-packed (column-major) weight tile.
__device__ __forceinline__ v16bf ldfragB(const __bf16* base, int hs) {
  return *(const v16bf*)(base + (hs << 4));
}
__device__ __forceinline__ v8f wmma_bf16(v16bf a, v16bf b, v8f c) {
  return __builtin_amdgcn_wmma_f32_16x16x32_bf16(false, a, false, b, (short)0, c, false, false);
}

// ---------------------------------------------------------------------------
// Generic tiled WMMA GEMM: C[Mrows,N] (+)= act(A[Mrows,K] @ W[K,N] + bias)
// 64x64 tile / block, 256 threads = 8 waves (4 M-subtiles x 2 N-subtiles).
// conv_dil>0: A is v[B*T,E] bf16 and logical K=2E, with the upper half
// reading v[t-dil] (zero-padded) -> causal dilated conv as a single GEMM.
// conv_w: W is the conv weight (O,E,2) f32, tap1 for k<E, tap0 for k>=E.
// Assumes Mrows%64==0, N%64==0, K%32==0 (true for every call here).
// ---------------------------------------------------------------------------
__global__ __launch_bounds__(256) void wmma_gemm(
    const void* __restrict__ A, int a_bf16, int lda,
    const void* __restrict__ W, int w_srow, int w_scol,
    const float* __restrict__ bias,
    void* __restrict__ C, int c_bf16, int ldc,
    int K, int act, int accum, int conv_dil, int conv_w)
{
  __shared__ __align__(32) __bf16 As[64][32];
  __shared__ __align__(32) __bf16 Ws[64][32];
  const int tid = threadIdx.x;
  const int wave = tid >> 5, lane = tid & 31;
  const int wm = wave & 3, wn = wave >> 2;
  const int l = lane & 15, hs = lane >> 4;
  const int m0 = blockIdx.y * 64, n0 = blockIdx.x * 64;
  v8f c0 = {}, c1 = {};

  for (int k0 = 0; k0 < K; k0 += 32) {
    for (int i = tid; i < 64 * 32; i += 256) {
      int m = i >> 5, k = i & 31;
      int row = m0 + m, col = k0 + k;
      float v;
      if (conv_dil == 0) {
        v = a_bf16 ? (float)((const __bf16*)A)[(size_t)row * lda + col]
                   : ((const float*)A)[(size_t)row * lda + col];
      } else {
        if (col < EE) {
          v = (float)((const __bf16*)A)[(size_t)row * EE + col];
        } else {
          int t = row & (TT - 1);
          v = (t >= conv_dil)
                ? (float)((const __bf16*)A)[(size_t)(row - conv_dil) * EE + (col - EE)]
                : 0.f;
        }
      }
      As[m][k] = (__bf16)v;
    }
    for (int i = tid; i < 64 * 32; i += 256) {
      int n = i >> 5, k = i & 31;
      int kk = k0 + k, nn = n0 + n;
      float v;
      if (conv_w) {
        const float* Wf = (const float*)W;
        v = (kk < EE) ? Wf[(size_t)nn * (EE * 2) + kk * 2 + 1]
                      : Wf[(size_t)nn * (EE * 2) + (kk - EE) * 2 + 0];
      } else {
        v = ((const float*)W)[(size_t)kk * w_srow + (size_t)nn * w_scol];
      }
      Ws[n][k] = (__bf16)v;
    }
    __syncthreads();
    v16bf a  = ldfragA(&As[wm * 16 + l][0], hs);
    v16bf b0 = ldfragB(&Ws[wn * 32 + l][0], hs);
    v16bf b1 = ldfragB(&Ws[wn * 32 + 16 + l][0], hs);
    c0 = wmma_bf16(a, b0, c0);
    c1 = wmma_bf16(a, b1, c1);
    __syncthreads();
  }

  for (int j = 0; j < 2; ++j) {
    v8f cc = j ? c1 : c0;
    int coln = n0 + wn * 32 + j * 16 + l;
    float bv = bias ? bias[coln] : 0.f;
#pragma unroll
    for (int v = 0; v < 8; ++v) {
      int rowm = m0 + wm * 16 + hs * 8 + v;   // C layout: lanes0-15 M=v, 16-31 M=v+8
      float val = cc[v] + bv;
      if (act == 1) val = fmaxf(val, 0.f);
      else if (act == 2) val = tanhf(val);
      else if (act == 3) val = sigf(val);
      size_t o = (size_t)rowm * ldc + coln;
      if (c_bf16) {
        __bf16* Cb = (__bf16*)C;
        Cb[o] = accum ? (__bf16)((float)Cb[o] + val) : (__bf16)val;
      } else {
        float* Cf = (float*)C;
        if (accum) Cf[o] += val; else Cf[o] = val;
      }
    }
  }
}

// ---------------------------------------------------------------------------
// GRU-D (GRU-B) recurrent scan. One block = 16 batch rows, 16 waves.
// pre[B,T,768] = d@W_d + m@W_m + b (z|r|h), precomputed by WMMA GEMMs.
// Weights are bf16 packed [N][K] so B-frags are two 16B L2 loads.
// ---------------------------------------------------------------------------
__global__ __launch_bounds__(512) void grub_scan(
    const float* __restrict__ x, const float* __restrict__ Mmask,
    const __bf16* __restrict__ D,
    const __bf16* __restrict__ pre,
    const __bf16* __restrict__ wfhT,                                  // [128][256]
    const __bf16* __restrict__ wxzT, const __bf16* __restrict__ wxrT,
    const __bf16* __restrict__ wxhT,                                  // [256][128]
    const __bf16* __restrict__ whzT, const __bf16* __restrict__ whrT,
    const __bf16* __restrict__ whhT,                                  // [256][256]
    const float* __restrict__ bfh,
    __bf16* __restrict__ out, int rev, int outoff)
{
  __shared__ __align__(32) __bf16 h_s[16][256];
  __shared__ __align__(32) __bf16 xh_s[16][128];
  __shared__ __align__(32) __bf16 x1_s[16][128];
  __shared__ __align__(32) __bf16 z_s[16][256];
  __shared__ __align__(32) __bf16 rh_s[16][256];
  __shared__ __align__(32) __bf16 ht_s[16][256];
  const int tid = threadIdx.x, wave = tid >> 5, lane = tid & 31;
  const int l = lane & 15, hs = lane >> 4;
  const int b0 = blockIdx.x * 16;

  for (int i = tid; i < 16 * 256; i += 512) ((__bf16*)h_s)[i] = (__bf16)0.f;
  __syncthreads();

  for (int t = 0; t < TT; ++t) {
    const int ti = rev ? (TT - 1 - t) : t;
    // 1) x_hat = sigmoid(h @ wfh + bfh): 8 N-tiles on waves 0..7
    if (wave < 8) {
      v8f acc = {};
      const int n0 = wave * 16;
#pragma unroll
      for (int k0 = 0; k0 < 256; k0 += 32) {
        acc = wmma_bf16(ldfragA(&h_s[l][k0], hs),
                        ldfragB(&wfhT[(size_t)(n0 + l) * 256 + k0], hs), acc);
      }
      const int col = n0 + l;
      const float bv = bfh[col];
#pragma unroll
      for (int v = 0; v < 8; ++v) xh_s[hs * 8 + v][col] = (__bf16)sigf(acc[v] + bv);
    }
    __syncthreads();
    // 2) x1 = fst ? x : m*x + (1-m)*x_hat*d
    for (int i = tid; i < 16 * 128; i += 512) {
      int r = i >> 7, f = i & 127;
      size_t g = ((size_t)(b0 + r) * TT + ti) * FF + f;
      float xv = x[g], mv = Mmask[g], dv = (float)D[g];
      float v = (t == 0) ? xv : (mv * xv + (1.f - mv) * (float)xh_s[r][f] * dv);
      x1_s[r][f] = (__bf16)v;
    }
    __syncthreads();
    // 3) z and r gates: each wave does its 16-col tile of z then r
    for (int j = 0; j < 2; ++j) {
      const int isr = j;
      const int n0 = wave * 16;
      const __bf16* WxT = isr ? wxrT : wxzT;
      const __bf16* WhT = isr ? whrT : whzT;
      const int poff = isr ? 256 : 0;
      v8f acc = {};
#pragma unroll
      for (int k0 = 0; k0 < 128; k0 += 32)
        acc = wmma_bf16(ldfragA(&x1_s[l][k0], hs),
                        ldfragB(&WxT[(size_t)(n0 + l) * 128 + k0], hs), acc);
#pragma unroll
      for (int k0 = 0; k0 < 256; k0 += 32)
        acc = wmma_bf16(ldfragA(&h_s[l][k0], hs),
                        ldfragB(&WhT[(size_t)(n0 + l) * 256 + k0], hs), acc);
      const int col = n0 + l;
#pragma unroll
      for (int v = 0; v < 8; ++v) {
        int r = hs * 8 + v;
        float p = (float)pre[((size_t)(b0 + r) * TT + ti) * 768 + poff + col];
        float s = sigf(acc[v] + p);
        if (isr) rh_s[r][col] = (__bf16)(s * (float)h_s[r][col]);
        else     z_s[r][col]  = (__bf16)s;
      }
    }
    __syncthreads();
    // 4) h_til = tanh(x1@w_xh + (r*h)@w_hh + pre_h)
    {
      const int n0 = wave * 16;
      v8f acc = {};
#pragma unroll
      for (int k0 = 0; k0 < 128; k0 += 32)
        acc = wmma_bf16(ldfragA(&x1_s[l][k0], hs),
                        ldfragB(&wxhT[(size_t)(n0 + l) * 128 + k0], hs), acc);
#pragma unroll
      for (int k0 = 0; k0 < 256; k0 += 32)
        acc = wmma_bf16(ldfragA(&rh_s[l][k0], hs),
                        ldfragB(&whhT[(size_t)(n0 + l) * 256 + k0], hs), acc);
      const int col = n0 + l;
#pragma unroll
      for (int v = 0; v < 8; ++v) {
        int r = hs * 8 + v;
        float p = (float)pre[((size_t)(b0 + r) * TT + ti) * 768 + 512 + col];
        ht_s[r][col] = (__bf16)tanhf(acc[v] + p);
      }
    }
    __syncthreads();
    // 5) h = (1-z)*h + z*h_til ; store (reverse dir writes reversed slot)
    for (int i = tid; i < 16 * 256; i += 512) {
      int r = i >> 8, n = i & 255;
      float hn = (1.f - (float)z_s[r][n]) * (float)h_s[r][n]
               + (float)z_s[r][n] * (float)ht_s[r][n];
      h_s[r][n] = (__bf16)hn;
      out[((size_t)(b0 + r) * TT + ti) * 512 + outoff + n] = (__bf16)hn;
    }
    __syncthreads();
  }
}

// ---------------------------------------------------------------------------
// Standard GRU scan (one direction). gi[B,T,768] = x@w_ih + b_ih (r|z|n).
// ---------------------------------------------------------------------------
__global__ __launch_bounds__(512) void gru_scan(
    const __bf16* __restrict__ gi, const __bf16* __restrict__ whhT, // [768][256]
    const float* __restrict__ bhh,
    __bf16* __restrict__ out, int rev, int outoff)
{
  __shared__ __align__(32) __bf16 h_s[16][256];
  __shared__ __align__(32) __bf16 gh_s[16][768];
  const int tid = threadIdx.x, wave = tid >> 5, lane = tid & 31;
  const int l = lane & 15, hs = lane >> 4;
  const int b0 = blockIdx.x * 16;
  for (int i = tid; i < 16 * 256; i += 512) ((__bf16*)h_s)[i] = (__bf16)0.f;
  __syncthreads();
  for (int t = 0; t < TT; ++t) {
    const int ti = rev ? (TT - 1 - t) : t;
    for (int j = 0; j < 3; ++j) {                 // 48 gh tiles / 16 waves
      const int n0 = (wave + j * 16) * 16;
      v8f acc = {};
#pragma unroll
      for (int k0 = 0; k0 < 256; k0 += 32)
        acc = wmma_bf16(ldfragA(&h_s[l][k0], hs),
                        ldfragB(&whhT[(size_t)(n0 + l) * 256 + k0], hs), acc);
      const int col = n0 + l;
      const float bv = bhh[col];
#pragma unroll
      for (int v = 0; v < 8; ++v) gh_s[hs * 8 + v][col] = (__bf16)(acc[v] + bv);
    }
    __syncthreads();
    for (int i = tid; i < 16 * 256; i += 512) {
      int r = i >> 8, n = i & 255;
      size_t gb = ((size_t)(b0 + r) * TT + ti) * 768;
      float rr = sigf((float)gi[gb + n]        + (float)gh_s[r][n]);
      float zz = sigf((float)gi[gb + 256 + n]  + (float)gh_s[r][256 + n]);
      float nn = tanhf((float)gi[gb + 512 + n] + rr * (float)gh_s[r][512 + n]);
      float hn = (1.f - zz) * nn + zz * (float)h_s[r][n];
      h_s[r][n] = (__bf16)hn;
      out[((size_t)(b0 + r) * TT + ti) * 512 + outoff + n] = (__bf16)hn;
    }
    __syncthreads();
  }
}

// ---------------------------------------------------------------------------
// Small elementwise / pack / reduction kernels
// ---------------------------------------------------------------------------
__global__ void dcalc_kernel(const float* __restrict__ Delta, __bf16* __restrict__ D, int n) {
  int i = blockIdx.x * blockDim.x + threadIdx.x;
  if (i < n) D[i] = (__bf16)(1.f - sigf(Delta[i]));
}
// dst[n*K + k] = (bf16) src[k*srow + n*scol]   (pack weights [N][K] bf16)
__global__ void pack_kernel(const float* __restrict__ src, __bf16* __restrict__ dst,
                            int K, int N, int srow, int scol) {
  int i = blockIdx.x * blockDim.x + threadIdx.x;
  if (i >= K * N) return;
  int n = i / K, k = i - n * K;
  dst[(size_t)n * K + k] = (__bf16)src[(size_t)k * srow + (size_t)n * scol];
}
__global__ void prod_kernel(const __bf16* __restrict__ a, const __bf16* __restrict__ b,
                            __bf16* __restrict__ c, int n) {
  int i = blockIdx.x * blockDim.x + threadIdx.x;
  if (i < n) c[i] = (__bf16)((float)a[i] * (float)b[i]);
}
// out3[row] = sigmoid(dot(wcout[row,:], w_pred) + b): one wave per row
__global__ __launch_bounds__(256) void pred_kernel(const __bf16* __restrict__ wcout,
                                                   const float* __restrict__ w,
                                                   const float* __restrict__ b,
                                                   float* __restrict__ out3) {
  int row = blockIdx.x * 8 + (threadIdx.x >> 5);
  int lane = threadIdx.x & 31;
  float s = 0.f;
  for (int k = lane; k < HH; k += 32) s += (float)wcout[(size_t)row * HH + k] * w[k];
  for (int off = 16; off; off >>= 1) s += __shfl_xor(s, off, 32);
  if (lane == 0) out3[row] = sigf(s + b[0]);
}

// ---------------------------------------------------------------------------
// Host-side orchestration
// ---------------------------------------------------------------------------
// d_in layout: [x, M, Delta, lens] then params flattened with dict keys
// sorted alphabetically at each level (jax tree order).
enum {
  IN_X = 0, IN_M = 1, IN_DELTA = 2, IN_LENS = 3,
  C1_B = 4, C1_W = 5, C2_B = 6, C2_W = 7, C3_B = 8, C3_W = 9, C4_B = 10, C4_W = 11,
  EMB_B = 12, EMB_W = 13,
  G1 = 14, G2 = 31,   // 17-entry gru_b blocks:
  // +0 b_h, +1 b_r, +2 b_z, +3 bfh, +4 w_dh, +5 w_dr, +6 w_dz,
  // +7 w_hh, +8 w_hr, +9 w_hz, +10 w_mh, +11 w_mr, +12 w_mz,
  // +13 w_xh, +14 w_xr, +15 w_xz, +16 wfh
  H2X_B = 48, H2X_W = 49, PRED_B = 50, PRED_W = 51,
  RNB_BHH = 52, RNB_BIH = 53, RNB_WHH = 54, RNB_WIH = 55,
  RNF_BHH = 56, RNF_BIH = 57, RNF_WHH = 58, RNF_WIH = 59,
  WC_B = 60, WC_W = 61
};

extern "C" void kernel_launch(void* const* d_in, const int* in_sizes, int n_in,
                              void* d_out, int out_size, void* d_ws, size_t ws_size,
                              hipStream_t stream) {
  (void)in_sizes; (void)n_in; (void)out_size; (void)ws_size;
  auto P = [&](int i) { return (const float*)d_in[i]; };

  // workspace carve-out (256B aligned)
  uint8_t* ws = (uint8_t*)d_ws;
  size_t cur = 0;
  auto take = [&](size_t bytes) {
    void* p = ws + cur;
    cur = (cur + bytes + 255) & ~(size_t)255;
    return p;
  };
  __bf16* bD   = (__bf16*)take((size_t)NBT * FF * 2);    // 1 - sigmoid(Delta)
  __bf16* bBig = (__bf16*)take((size_t)NBT * 768 * 2);   // pre-gates / gi / prod
  __bf16* bHbi = (__bf16*)take((size_t)NBT * 512 * 2);   // h_bi, reused as wc_out
  __bf16* bV   = (__bf16*)take((size_t)NBT * 256 * 2);   // embedding v
  __bf16* bOut = (__bf16*)take((size_t)NBT * 512 * 2);   // bi-GRU outs
  __bf16* bTot = (__bf16*)take((size_t)NBT * 512 * 2);   // conv total
  __bf16 *pk_wfh[2], *pk_wx[2][3], *pk_wh[2][3];
  for (int d = 0; d < 2; ++d) {
    pk_wfh[d] = (__bf16*)take(128 * 256 * 2);
    for (int g = 0; g < 3; ++g) pk_wx[d][g] = (__bf16*)take(256 * 128 * 2);
    for (int g = 0; g < 3; ++g) pk_wh[d][g] = (__bf16*)take(256 * 256 * 2);
  }
  __bf16* pk_rnnf = (__bf16*)take(768 * 256 * 2);
  __bf16* pk_rnnb = (__bf16*)take(768 * 256 * 2);

  float* x_preds = (float*)d_out + NBT;  // outputs: [out3 | x_preds]

  auto gemm = [&](const void* A, int a_bf16, int lda,
                  const void* W, int w_srow, int w_scol,
                  const float* bias, void* C, int c_bf16, int ldc,
                  int N, int K, int act, int accum, int conv_dil, int conv_w) {
    dim3 grid(N / 64, NBT / 64);
    wmma_gemm<<<grid, 256, 0, stream>>>(A, a_bf16, lda, W, w_srow, w_scol,
                                        bias, C, c_bf16, ldc, K, act, accum,
                                        conv_dil, conv_w);
  };
  auto pack = [&](const float* src, __bf16* dst, int K, int N, int srow, int scol) {
    pack_kernel<<<(K * N + 255) / 256, 256, 0, stream>>>(src, dst, K, N, srow, scol);
  };

  // 0) decay
  dcalc_kernel<<<(NBT * FF) / 256, 256, 0, stream>>>(P(IN_DELTA), bD, NBT * FF);

  // 1) pack all in-scan weights to bf16 [N][K]
  for (int d = 0; d < 2; ++d) {
    int base = d ? G2 : G1;
    pack(P(base + 16), pk_wfh[d], 256, 128, 128, 1);                 // wfh (H,F)
    pack(P(base + 15), pk_wx[d][0], 128, 256, 256, 1);               // w_xz
    pack(P(base + 14), pk_wx[d][1], 128, 256, 256, 1);               // w_xr
    pack(P(base + 13), pk_wx[d][2], 128, 256, 256, 1);               // w_xh
    pack(P(base + 9),  pk_wh[d][0], 256, 256, 256, 1);               // w_hz
    pack(P(base + 8),  pk_wh[d][1], 256, 256, 256, 1);               // w_hr
    pack(P(base + 7),  pk_wh[d][2], 256, 256, 256, 1);               // w_hh
  }
  pack(P(RNF_WHH), pk_rnnf, 256, 768, 768, 1);
  pack(P(RNB_WHH), pk_rnnb, 256, 768, 768, 1);

  // 2) GRU-B scans (fwd then rev), pre-gates hoisted into big GEMMs
  const int wd[3] = {6, 5, 4}, wm[3] = {12, 11, 10}, bb[3] = {2, 1, 0}; // z,r,h
  for (int d = 0; d < 2; ++d) {
    int base = d ? G2 : G1;
    for (int g = 0; g < 3; ++g) {
      gemm(bD, 1, FF, P(base + wd[g]), HH, 1, P(base + bb[g]),
           bBig + g * 256, 1, 768, HH, FF, 0, 0, 0, 0);              // d@W_d + b
      gemm(P(IN_M), 0, FF, P(base + wm[g]), HH, 1, nullptr,
           bBig + g * 256, 1, 768, HH, FF, 0, 1, 0, 0);              // += m@W_m
    }
    grub_scan<<<BB / 16, 512, 0, stream>>>(
        P(IN_X), P(IN_M), bD, bBig, pk_wfh[d],
        pk_wx[d][0], pk_wx[d][1], pk_wx[d][2],
        pk_wh[d][0], pk_wh[d][1], pk_wh[d][2],
        P(base + 3), bHbi, d, d * HH);
  }

  // 3) x_preds = h_bi @ W_h2x + b  (f32, straight into d_out)
  gemm(bHbi, 1, 512, P(H2X_W), FF, 1, P(H2X_B), x_preds, 0, FF, FF, 512, 0, 0, 0, 0);
  // 4) v = relu(x_preds @ W_emb + b)
  gemm(x_preds, 0, FF, P(EMB_W), EE, 1, P(EMB_B), bV, 1, EE, EE, FF, 1, 0, 0, 0);

  // 5) bidirectional GRU: gi hoisted, then scans
  gemm(bV, 1, EE, P(RNF_WIH), 768, 1, P(RNF_BIH), bBig, 1, 768, 768, EE, 0, 0, 0, 0);
  gru_scan<<<BB / 16, 512, 0, stream>>>(bBig, pk_rnnf, P(RNF_BHH), bOut, 0, 0);
  gemm(bV, 1, EE, P(RNB_WIH), 768, 1, P(RNB_BIH), bBig, 1, 768, 768, EE, 0, 0, 0, 0);
  gru_scan<<<BB / 16, 512, 0, stream>>>(bBig, pk_rnnb, P(RNB_BHH), bOut, 1, HH);

  // 6) total = sum_d relu(causal_conv_d(v)); each conv = one K=2E WMMA GEMM
  const int dil[4] = {1, 2, 4, 8};
  const int cw[4] = {C1_W, C2_W, C3_W, C4_W}, cb[4] = {C1_B, C2_B, C3_B, C4_B};
  for (int i = 0; i < 4; ++i)
    gemm(bV, 1, EE, P(cw[i]), 0, 0, P(cb[i]), bTot, 1, 512,
         512, 2 * EE, /*relu*/1, /*accum*/i ? 1 : 0, dil[i], /*conv_w*/1);

  // 7) out3 = sigmoid(tanh((total*outs) @ W_wc + b) @ w_pred + b)
  prod_kernel<<<(NBT * 512) / 256, 256, 0, stream>>>(bTot, bOut, bBig, NBT * 512);
  gemm(bBig, 1, 512, P(WC_W), HH, 1, P(WC_B), bHbi, 1, HH, HH, 512, /*tanh*/2, 0, 0, 0);
  pred_kernel<<<NBT / 8, 256, 0, stream>>>(bHbi, P(PRED_W), P(PRED_B), (float*)d_out);
}
